// GraphFeatureExtraction_42640435315454
// MI455X (gfx1250) — compile-verified, matrix operand order checked by reading, not run on hardware
//
#include <hip/hip_runtime.h>

// ---------------------------------------------------------------------------
// GraphFeatureExtraction fused kernel for gfx1250 (MI455X).
//
// reference:  y = 0.5*(x@W_out+b_out) + 0.5*(x@W_in+b_in) + (x@W_root+b_root)
// fused:      y = x @ Wc + bc,  Wc = 0.5*W_out + 0.5*W_in + W_root
// GEMM:       M=16384 (B*N), N=512 (OUT), K=1024 (L), f32 in/out,
//             bf16 WMMA (v_wmma_f32_16x16x32_bf16) with f32 accumulation.
// B-tiles are DMA'd global->LDS by the Tensor Data Mover (tensor_load_to_lds)
// with hardware row padding; A-tiles are register-staged (need f32->bf16 cvt).
// ---------------------------------------------------------------------------

typedef __bf16 bf16;
typedef bf16  bf16x4  __attribute__((ext_vector_type(4)));
typedef bf16  bf16x8  __attribute__((ext_vector_type(8)));
typedef bf16  bf16x16 __attribute__((ext_vector_type(16)));
typedef float floatx4 __attribute__((ext_vector_type(4)));
typedef float floatx8 __attribute__((ext_vector_type(8)));
typedef unsigned int u32x4 __attribute__((ext_vector_type(4)));
typedef int          i32x4 __attribute__((ext_vector_type(4)));
typedef int          i32x8 __attribute__((ext_vector_type(8)));

constexpr int Mdim = 64 * 256;   // 16384 rows (B*N)
constexpr int Ndim = 512;        // OUT
constexpr int Kdim = 1024;       // L
constexpr int BM = 128;
constexpr int BN = 128;
constexpr int BK = 32;
constexpr int LDT = BK + 8;      // padded LDS row stride (80B) -> conflict-free b128

// ---------------------------------------------------------------------------
// Phase 1: fold 3 weight matrices -> bf16 transposed Wt[N][K]; combine biases.
// ---------------------------------------------------------------------------
__global__ __launch_bounds__(256)
void prep_weights(const float* __restrict__ W_in, const float* __restrict__ b_in,
                  const float* __restrict__ W_out, const float* __restrict__ b_out,
                  const float* __restrict__ W_root, const float* __restrict__ b_root,
                  bf16* __restrict__ Wt, float* __restrict__ bias) {
  const int idx = blockIdx.x * blockDim.x + threadIdx.x;   // 0 .. K*N-1
  const int k = idx >> 9;        // / Ndim
  const int n = idx & (Ndim - 1);
  const float w = 0.5f * W_out[idx] + 0.5f * W_in[idx] + W_root[idx];
  Wt[n * Kdim + k] = (bf16)w;    // transpose: Wt is [N][K], K-contiguous
  if (idx < Ndim) {
    bias[idx] = 0.5f * b_out[idx] + 0.5f * b_in[idx] + b_root[idx];
  }
}

// ---------------------------------------------------------------------------
// TDM: DMA one B tile (BN rows x BK bf16, row-major K-contiguous) from Wt into
// LDS at lds_byte_off, inserting 16B of padding after every 64B row so the
// LDS image has the same 80B row stride the fragment loader expects.
// D# layout per CDNA5 ISA.
// Toolchain note: this clang exposes the 6-arg builtin
//   (u32x4 g0, i32x8 g1, i32x4 g2, i32x4 g3, i32x8 extra, i32 cpol);
// groups 2/3 and the trailing i32x8 are zero (2D tile, no extra descriptor).
// ---------------------------------------------------------------------------
__device__ __forceinline__ void tdm_load_Btile(unsigned int lds_byte_off,
                                               const bf16* gtile) {
  const unsigned long long ga = (unsigned long long)(uintptr_t)gtile;

  u32x4 g0;
  g0.x = 1u;                                   // count=1 (valid user descriptor)
  g0.y = lds_byte_off;                         // D#.lds_addr
  g0.z = (unsigned int)ga;                     // global_addr[31:0]
  g0.w = (unsigned int)(ga >> 32) | (2u << 30);// global_addr[56:32] | type=2

  i32x8 g1;
  g1.s0 = (int)((1u << 16)     // data_size = 2 bytes
              | (1u << 20)     // pad_enable
              | (3u << 22)     // pad_interval: 16 DWORDs (64B) between pads
              | (3u << 25));   // pad_amount:   4 DWORDs (16B) of padding
  g1.s1 = (int)((unsigned)Kdim << 16); // tensor_dim0[15:0] in bits[63:48]
  g1.s2 = (int)((unsigned)Ndim << 16); // tensor_dim0 hi=0 | tensor_dim1[15:0]
  g1.s3 = (int)((unsigned)BK  << 16);  // tensor_dim1 hi=0 | tile_dim0=32
  g1.s4 = BN;                          // tile_dim1=128 | tile_dim2=0
  g1.s5 = Kdim;                        // tensor_dim0_stride[31:0] = 1024
  g1.s6 = 0;                           // td0_stride hi16 | td1_stride lo16
  g1.s7 = (int)(((unsigned long long)Kdim * Ndim) >> 16); // td1_stride hi32

  i32x4 gz4 = {0, 0, 0, 0};            // groups 2/3: dims beyond 2D unused
  i32x8 gz8 = {0, 0, 0, 0, 0, 0, 0, 0};
  __builtin_amdgcn_tensor_load_to_lds(g0, g1, gz4, gz4, gz8, 0 /*cpol*/);
}

// ---------------------------------------------------------------------------
// Fragment fetch from an LDS tile laid out [rows][LDT] bf16 (K-contiguous).
// ISA 16-bit A/B operand layout (wave32):
//   lanes 0-15 : row = lane,    halves 0-7 = K0..7,   halves 8-15 = K16..23
//   lanes 16-31: row = lane-16, halves 0-7 = K8..15,  halves 8-15 = K24..31
// ---------------------------------------------------------------------------
__device__ __forceinline__ bf16x16 frag_from_lds(const bf16* __restrict__ tile,
                                                 int row0, int lane) {
  const bf16* p = tile + (row0 + (lane & 15)) * LDT + ((lane >> 4) << 3);
  bf16x8 lo = *(const bf16x8*)(p);        // ds_load_b128
  bf16x8 hi = *(const bf16x8*)(p + 16);   // ds_load_b128
  return __builtin_shufflevector(lo, hi, 0, 1, 2, 3, 4, 5, 6, 7,
                                         8, 9, 10, 11, 12, 13, 14, 15);
}

// ---------------------------------------------------------------------------
// Phase 2: tiled bf16 WMMA GEMM.  Block = 8 wave32 (256 thr), tile 128x128,
// wave sub-tile 32x64 = 2x4 WMMA accumulators. Double-buffered LDS; A slabs
// register-prefetched, B slabs DMA'd by the TDM.
// ---------------------------------------------------------------------------
__global__ __launch_bounds__(256)
void gemm_bf16_wmma(const float* __restrict__ x, const bf16* __restrict__ Wt,
                    const float* __restrict__ bias, float* __restrict__ out) {
  __shared__ __align__(16) bf16 As[2][BM * LDT];   // 2 * 10240 B
  __shared__ __align__(16) bf16 Bs[2][BN * LDT];   // 2 * 10240 B  (N-major)

  const int tid  = threadIdx.x;
  const int lane = tid & 31;
  const int wave = tid >> 5;
  const int waveM = (wave & 3) * 32;   // 4 waves along M
  const int waveN = (wave >> 2) * 64;  // 2 waves along N
  const int blockM = blockIdx.y * BM;
  const int blockN = blockIdx.x * BN;

  // A staging: each thread loads 4x float4  (rows arow+32i, K cols acol..+3)
  const int arow = tid >> 3;          // 0..31
  const int acol = (tid & 7) << 2;    // 0,4,...,28

  floatx4 aReg[4];

  auto globalLoadA = [&](int k0) {
#pragma unroll
    for (int i = 0; i < 4; ++i) {
      const float* p = x + (blockM + arow + i * 32) * Kdim + k0 + acol;
      aReg[i] = *(const floatx4*)p;                       // global_load_b128
    }
  };

  auto ldsStoreA = [&](int buf) {
#pragma unroll
    for (int i = 0; i < 4; ++i) {
      bf16x4 c;
#pragma unroll
      for (int j = 0; j < 4; ++j) c[j] = (bf16)aReg[i][j];  // f32 -> bf16
      *(bf16x4*)&As[buf][(arow + i * 32) * LDT + acol] = c;
    }
  };

  auto bsOff = [&](int buf) -> unsigned int {
    return (unsigned int)(uintptr_t)(&Bs[buf][0]);   // low 32 bits = LDS addr
  };

  floatx8 acc[2][4] = {};

  // Prologue: slab 0
  if (wave == 0) tdm_load_Btile(bsOff(0), Wt + blockN * Kdim);
  globalLoadA(0);
  ldsStoreA(0);
  if (wave == 0) __builtin_amdgcn_s_wait_tensorcnt(0);
  __syncthreads();

  int buf = 0;
  constexpr int steps = Kdim / BK;   // 32
  for (int ks = 0; ks < steps; ++ks) {
    const bool hasNext = (ks + 1) < steps;
    if (hasNext) {
      // Issue next B slab DMA (target buffer's readers finished last iter).
      if (wave == 0)
        tdm_load_Btile(bsOff(buf ^ 1), Wt + blockN * Kdim + (ks + 1) * BK);
      globalLoadA((ks + 1) * BK);    // prefetch next A slab into registers
      if (ks + 2 < steps)            // speculative L2 warm-up two slabs ahead
        __builtin_prefetch(x + (blockM + arow) * Kdim + (ks + 2) * BK + acol, 0, 1);
    }

    bf16x16 aF[2], bF[4];
#pragma unroll
    for (int mt = 0; mt < 2; ++mt)
      aF[mt] = frag_from_lds(As[buf], waveM + mt * 16, lane);
#pragma unroll
    for (int nt = 0; nt < 4; ++nt)
      bF[nt] = frag_from_lds(Bs[buf], waveN + nt * 16, lane);

#pragma unroll
    for (int mt = 0; mt < 2; ++mt)
#pragma unroll
      for (int nt = 0; nt < 4; ++nt)
        acc[mt][nt] = __builtin_amdgcn_wmma_f32_16x16x32_bf16(
            false, aF[mt], false, bF[nt], (short)0, acc[mt][nt], false, false);

    if (hasNext) {
      __syncthreads();               // all waves done reading As[buf^1]
      ldsStoreA(buf ^ 1);
      if (wave == 0) __builtin_amdgcn_s_wait_tensorcnt(0);  // B slab landed
      __syncthreads();               // publish new slab to all waves
      buf ^= 1;
    }
  }

  // Epilogue: C/D layout -> VGPR r: lanes 0-15 hold M=r, lanes 16-31 hold M=8+r,
  // N = lane & 15. Add combined bias, store f32.
#pragma unroll
  for (int mt = 0; mt < 2; ++mt) {
#pragma unroll
    for (int nt = 0; nt < 4; ++nt) {
      const int col  = blockN + waveN + nt * 16 + (lane & 15);
      const int row0 = blockM + waveM + mt * 16 + ((lane >> 4) << 3);
      const float bv = bias[col];
      float* po = out + row0 * Ndim + col;
#pragma unroll
      for (int r = 0; r < 8; ++r) po[r * Ndim] = acc[mt][nt][r] + bv;
    }
  }
}

// ---------------------------------------------------------------------------
// Host-side launcher.  Inputs (setup_inputs order):
//   0:x f32[16384,1024] 1:At i32 (unused) 2:W_in 3:b_in 4:W_out 5:b_out
//   6:W_root 7:b_root.   Output: f32[16384,512].
// Workspace: Wt bf16[512*1024] (1 MB) + bias f32[512].
// ---------------------------------------------------------------------------
extern "C" void kernel_launch(void* const* d_in, const int* in_sizes, int n_in,
                              void* d_out, int out_size, void* d_ws, size_t ws_size,
                              hipStream_t stream) {
  const float* x      = (const float*)d_in[0];
  const float* W_in   = (const float*)d_in[2];
  const float* b_in   = (const float*)d_in[3];
  const float* W_out  = (const float*)d_in[4];
  const float* b_out  = (const float*)d_in[5];
  const float* W_root = (const float*)d_in[6];
  const float* b_root = (const float*)d_in[7];
  float* out = (float*)d_out;

  bf16*  Wt   = (bf16*)d_ws;
  float* bias = (float*)((char*)d_ws + (size_t)Ndim * Kdim * sizeof(bf16));

  prep_weights<<<(Kdim * Ndim) / 256, 256, 0, stream>>>(
      W_in, b_in, W_out, b_out, W_root, b_root, Wt, bias);

  dim3 grid(Ndim / BN, Mdim / BM);   // (4, 128)
  gemm_bf16_wmma<<<grid, 256, 0, stream>>>(x, Wt, bias, out);
}
